// FasterRCNN_6794638262890
// MI455X (gfx1250) — matrix operand order repaired and checked by reading, not compiled, not run on hardware
//
#include <hip/hip_runtime.h>
#include <hip/hip_bf16.h>

// ---------- CDNA5 WMMA types ----------
typedef __bf16 bf16_t;
typedef bf16_t v16bf __attribute__((ext_vector_type(16)));
typedef bf16_t v8bf  __attribute__((ext_vector_type(8)));
typedef float  v8f   __attribute__((ext_vector_type(8)));
typedef unsigned int u32x4 __attribute__((ext_vector_type(4)));
typedef int          i32x8 __attribute__((ext_vector_type(8)));
typedef int          i32x4 __attribute__((ext_vector_type(4)));

union V16 { v16bf v; v8bf h[2]; };

// float -> bf16 (round-to-nearest-even), stored as raw u16
static __device__ __forceinline__ unsigned short f2bf(float f) {
    unsigned u = __float_as_uint(f);
    unsigned r = (u + 0x7fffu + ((u >> 16) & 1u)) >> 16;
    return (unsigned short)r;
}

// Problem constants
#define IMGS   2
#define CIN    2048
#define COUT   512
#define HH     50
#define WWID   50
#define HP     52            // padded spatial
#define MTOT   5000          // IMGS*HH*WWID
#define NANCH  9
#define NBOX   22500         // HH*WWID*NANCH
#define POSTN  2000

#define KSTAGE 128                     // ci per staged B tile
#define STAGES (9 * (CIN / KSTAGE))    // 144

// Tensor Data Mover path (gfx1250): guarded so the file always compiles.
#if defined(__AMDGCN__) && __has_builtin(__builtin_amdgcn_tensor_load_to_lds) && \
    __has_builtin(__builtin_amdgcn_s_wait_tensorcnt)
#define USE_TDM 1
#else
#define USE_TDM 0
#endif

// ---------------------------------------------------------------------------
// 1) Pad + convert x (NCHW fp32) -> xpad (N, 52, 52, Ci) bf16, zero border
// ---------------------------------------------------------------------------
__global__ void pad_convert_x(const float* __restrict__ x,
                              unsigned short* __restrict__ xpad) {
    int t = blockIdx.x * blockDim.x + threadIdx.x;
    const int TOT = IMGS * HP * HP * CIN;
    if (t >= TOT) return;
    int ci = t & (CIN - 1);
    int t2 = t >> 11;           // /2048
    int xx = t2 % HP; int t3 = t2 / HP;
    int yy = t3 % HP; int n  = t3 / HP;
    float v = 0.f;
    if (yy >= 1 && yy <= HH && xx >= 1 && xx <= WWID)
        v = x[(((size_t)n * CIN + ci) * HH + (yy - 1)) * WWID + (xx - 1)];
    xpad[t] = f2bf(v);
}

// ---------------------------------------------------------------------------
// 2) Convert conv1_w (Co, Ci, 3, 3) fp32 -> wt (tap, Co, Ci) bf16
// ---------------------------------------------------------------------------
__global__ void convert_w(const float* __restrict__ w,
                          unsigned short* __restrict__ wt) {
    int t = blockIdx.x * blockDim.x + threadIdx.x;
    const int TOT = 9 * COUT * CIN;
    if (t >= TOT) return;
    int ci  = t & (CIN - 1);
    int t2  = t >> 11;
    int co  = t2 & (COUT - 1);
    int tap = t2 >> 9;
    wt[t] = f2bf(w[((size_t)co * CIN + ci) * 9 + tap]);
}

// ---------------------------------------------------------------------------
// TDM: async-load one 64(co) x 128(ci) bf16 tile (row stride CIN) into LDS.
// D# per CDNA5 ISA (2D tile; data_size=2B,
// tensor_dim0=2048, tensor_dim1=64, tile=128x64, dim0_stride=2048).
// 6-arg toolchain form: (g0, g1, g2, g3, g_extra, cpol).
// ---------------------------------------------------------------------------
#if USE_TDM
static __device__ __forceinline__ void tdm_load_b_tile(const unsigned short* gsrc,
                                                       unsigned ldsoff) {
    unsigned long long ga = (unsigned long long)(size_t)gsrc;
    u32x4 g0;
    g0[0] = 1u;                                    // count=1 (valid user D#)
    g0[1] = ldsoff;                                // lds_addr (bytes)
    g0[2] = (unsigned)(ga & 0xffffffffu);          // global_addr[31:0]
    g0[3] = (unsigned)((ga >> 32) & 0x01ffffffu)   // global_addr[56:32]
            | (2u << 30);                          // type = 2 ("image")
    i32x8 g1;
    g1[0] = (1 << 16);                 // workgroup_mask=0, data_size=1 (2B)
    g1[1] = (int)((unsigned)CIN << 16);// tensor_dim0[15:0]=2048 in [31:16]
    g1[2] = (64 << 16);                // tensor_dim0[31:16]=0 | tensor_dim1 lo=64
    g1[3] = (KSTAGE << 16);            // tensor_dim1 hi=0 | tile_dim0=128
    g1[4] = 64;                        // tile_dim1=64, tile_dim2=0
    g1[5] = CIN;                       // tensor_dim0_stride lo32 = 2048
    g1[6] = 0;                         // stride hi16 | tensor_dim1_stride lo16
    g1[7] = 0;                         // tensor_dim1_stride hi32
    i32x4 z4 = {0, 0, 0, 0};
    i32x8 z8 = {0, 0, 0, 0, 0, 0, 0, 0};
    __builtin_amdgcn_tensor_load_to_lds(g0, g1, z4, z4, z8, 0);
}
#else
static __device__ __forceinline__ void coop_load_b_tile(const unsigned short* gsrc,
                                                        unsigned short* lds, int tid) {
    // 64 rows x 256B each = 1024 16B chunks; 256 threads x 4 chunks
    #pragma unroll
    for (int q = 0; q < 4; ++q) {
        int idx = q * 256 + tid;
        int row = idx >> 4;            // 16 chunks per row
        int seg = idx & 15;
        *(v8bf*)(lds + (size_t)row * KSTAGE + seg * 8) =
            *(const v8bf*)(gsrc + (size_t)row * CIN + seg * 8);
    }
}
#endif

// ---------------------------------------------------------------------------
// 3) 3x3 conv as implicit GEMM with v_wmma_f32_16x16x32_bf16.
//    M = (n,y,x) = 5000, N = Co = 512, K = Ci*9.
//    Block = 256 threads = 8 waves; each wave -> 16(M) x 64(N) strip.
//    B tiles (shared by all 8 waves) are TDM-staged into double-buffered LDS;
//    A fragments stream from global (per-wave) with prefetch.
// ---------------------------------------------------------------------------
__global__ void __launch_bounds__(256) conv3x3_wmma(
        const unsigned short* __restrict__ xpad,   // [n][52][52][Ci] bf16
        const unsigned short* __restrict__ wt,     // [tap][Co][Ci]   bf16
        const float* __restrict__ bias,            // [Co]
        float* __restrict__ hout)                  // [M][Co] fp32 (ReLU'd)
{
    __shared__ __align__(16) unsigned short btile[2][64 * KSTAGE]; // 32 KB

    const int tid  = threadIdx.x;
    const int lane = tid & 31;
    const int wave = tid >> 5;
    const int m0 = (blockIdx.x * 8 + wave) * 16;
    const int n0 = blockIdx.y * 64;

    // A-fragment row for this lane (16-bit A 16x32 layout)
    const int arow = m0 + (lane & 15);
    const int rc   = arow < MTOT ? arow : MTOT - 1;
    const int img  = rc / (HH * WWID);
    const int pix  = rc % (HH * WWID);
    const int oy   = pix / WWID, ox = pix % WWID;
    const size_t xbase = (((size_t)img * HP + oy) * HP + ox) * CIN;

    const int kbase = (lane >> 4) << 4;   // A: K sub-offset 0 or 16
    const int bcol  = lane & 15;          // B: N column within 16-wide subtile
    const int kgrp  = (lane >> 4) << 4;   // B: K group 0 or 16

    v8f acc0 = {}, acc1 = {}, acc2 = {}, acc3 = {};

    // ---- stage 0 B tile ----
    {
        const unsigned short* src0 = wt + (size_t)n0 * CIN;  // tap=0, c0=0
#if USE_TDM
        if (wave == 0)
            tdm_load_b_tile(src0, (unsigned)(size_t)&btile[0][0]);
#else
        coop_load_b_tile(src0, &btile[0][0], tid);
#endif
    }

    for (int kk = 0; kk < STAGES; ++kk) {
        const int tap = kk >> 4;
        const int c0  = (kk & 15) << 7;   // * KSTAGE
        const int buf = kk & 1;

        // issue next stage into the other buffer (safe: last read of that
        // buffer was fenced by the trailing __syncthreads of iteration kk-1)
        if (kk + 1 < STAGES) {
            const int ntap = (kk + 1) >> 4;
            const int nc0  = ((kk + 1) & 15) << 7;
            const unsigned short* nsrc = wt + ((size_t)ntap * COUT + n0) * CIN + nc0;
#if USE_TDM
            if (wave == 0)
                tdm_load_b_tile(nsrc, (unsigned)(size_t)&btile[buf ^ 1][0]);
            if (wave == 0) __builtin_amdgcn_s_wait_tensorcnt(1);
#else
            coop_load_b_tile(nsrc, &btile[buf ^ 1][0], tid);
#endif
        } else {
#if USE_TDM
            if (wave == 0) __builtin_amdgcn_s_wait_tensorcnt(0);
#endif
        }
        __syncthreads();   // stage kk resident for every wave

        const int ky = tap / 3, kx = tap - ky * 3;
        const bf16_t* xa = (const bf16_t*)xpad + xbase + ((size_t)ky * HP + kx) * CIN + c0;
        const unsigned short* lrow = &btile[buf][(size_t)bcol * KSTAGE];

        #pragma unroll
        for (int sub = 0; sub < 4; ++sub) {
            const int cc = sub * 32;
            V16 a;
            a.h[0] = *(const v8bf*)(xa + cc + kbase);
            a.h[1] = *(const v8bf*)(xa + cc + 8 + kbase);
            __builtin_prefetch((const void*)(xa + cc + 512), 0, 1);

            const unsigned short* lb = lrow + cc + kgrp;
            V16 b0, b1, b2, b3;
            b0.h[0] = *(const v8bf*)(lb);
            b0.h[1] = *(const v8bf*)(lb + 8);
            b1.h[0] = *(const v8bf*)(lb + 16 * KSTAGE);
            b1.h[1] = *(const v8bf*)(lb + 16 * KSTAGE + 8);
            b2.h[0] = *(const v8bf*)(lb + 32 * KSTAGE);
            b2.h[1] = *(const v8bf*)(lb + 32 * KSTAGE + 8);
            b3.h[0] = *(const v8bf*)(lb + 48 * KSTAGE);
            b3.h[1] = *(const v8bf*)(lb + 48 * KSTAGE + 8);

            acc0 = __builtin_amdgcn_wmma_f32_16x16x32_bf16(false, a.v, false, b0.v,
                                                           (short)0, acc0, false, false);
            acc1 = __builtin_amdgcn_wmma_f32_16x16x32_bf16(false, a.v, false, b1.v,
                                                           (short)0, acc1, false, false);
            acc2 = __builtin_amdgcn_wmma_f32_16x16x32_bf16(false, a.v, false, b2.v,
                                                           (short)0, acc2, false, false);
            acc3 = __builtin_amdgcn_wmma_f32_16x16x32_bf16(false, a.v, false, b3.v,
                                                           (short)0, acc3, false, false);
        }
        __syncthreads();   // all waves done reading buf before it is reloaded
    }

    // Epilogue: D layout — lane l, VGPR r -> M = r + (l>=16)*8, N = l&15
    const int rowoff = (lane >> 4) << 3;
    const float bv0 = bias[n0 + bcol];
    const float bv1 = bias[n0 + 16 + bcol];
    const float bv2 = bias[n0 + 32 + bcol];
    const float bv3 = bias[n0 + 48 + bcol];
    #pragma unroll
    for (int r = 0; r < 8; ++r) {
        int m = m0 + rowoff + r;
        if (m < MTOT) {
            float* o = hout + (size_t)m * COUT + n0 + bcol;
            o[0]  = fmaxf(acc0[r] + bv0, 0.f);
            o[16] = fmaxf(acc1[r] + bv1, 0.f);
            o[32] = fmaxf(acc2[r] + bv2, 0.f);
            o[48] = fmaxf(acc3[r] + bv3, 0.f);
        }
    }
}

// ---------------------------------------------------------------------------
// 4) 1x1 heads + softmax-fg (LDS-staged h row)
// ---------------------------------------------------------------------------
__global__ void __launch_bounds__(64) heads_kernel(
        const float* __restrict__ hout,
        const float* __restrict__ sw, const float* __restrict__ sb,
        const float* __restrict__ bw, const float* __restrict__ bb,
        float* __restrict__ scores,   // [n][22500] fg prob
        float* __restrict__ boxes)    // [n][22500][4]
{
    __shared__ float hrow[COUT];
    const int pos = blockIdx.x;
    const int tid = threadIdx.x;
    for (int i = tid; i < COUT; i += 64) hrow[i] = hout[(size_t)pos * COUT + i];
    __syncthreads();

    const int n = pos / (HH * WWID), pix = pos % (HH * WWID);
    if (tid < NANCH) {
        const float* w0 = sw + (size_t)(2 * tid) * COUT;
        const float* w1 = w0 + COUT;
        float s0 = sb[2 * tid], s1 = sb[2 * tid + 1];
        for (int c = 0; c < COUT; ++c) { float h = hrow[c]; s0 += h * w0[c]; s1 += h * w1[c]; }
        scores[(size_t)n * NBOX + pix * NANCH + tid] = 1.f / (1.f + __expf(s0 - s1));
    } else if (tid < NANCH + 36) {
        const int o = tid - NANCH;
        const float* w = bw + (size_t)o * COUT;
        float s = bb[o];
        for (int c = 0; c < COUT; ++c) s += hrow[c] * w[c];
        const int a = o >> 2, comp = o & 3;
        boxes[((size_t)n * NBOX + pix * NANCH + a) * 4 + comp] = s;
    }
}

// ---------------------------------------------------------------------------
// 5) Fused argmax-NMS (one 1024-thread block per image)
// ---------------------------------------------------------------------------
#define CE 22   // ceil(22500/1024)

__global__ void __launch_bounds__(1024) nms_kernel(
        const float* __restrict__ scores,
        const float* __restrict__ boxes,
        float* __restrict__ out)           // [IMGS*2000][5]
{
    const int img = blockIdx.x;
    const int tid = threadIdx.x;
    const float* sc = scores + (size_t)img * NBOX;
    const float* bx = boxes  + (size_t)img * NBOX * 4;

    float s[CE];
    unsigned valid = 0u;
    #pragma unroll
    for (int j = 0; j < CE; ++j) {
        int i = j * 1024 + tid;
        bool in = i < NBOX;
        s[j] = in ? sc[i] : -1e30f;
        if (in) valid |= (1u << j);
    }

    __shared__ float rv[1024];
    __shared__ int   ri[1024];
    __shared__ float cbox[4];
    __shared__ int   csel;

    for (int k = 0; k < POSTN; ++k) {
        float best = -1e30f; int bi = -1;
        unsigned v0 = valid;
        while (v0) {
            int j = __ffs(v0) - 1;
            v0 &= v0 - 1;
            if (s[j] > best) { best = s[j]; bi = j * 1024 + tid; }
        }
        rv[tid] = best; ri[tid] = bi;
        __syncthreads();
        for (int off = 512; off > 0; off >>= 1) {
            if (tid < off) {
                if (rv[tid + off] > rv[tid]) { rv[tid] = rv[tid + off]; ri[tid] = ri[tid + off]; }
            }
            __syncthreads();
        }
        if (tid == 0) {
            int idx = ri[0];
            csel = idx;
            float b0 = 0.f, b1 = 0.f, b2 = 0.f, b3 = 0.f;
            if (idx >= 0) {
                const float* p = bx + (size_t)idx * 4;
                b0 = p[0]; b1 = p[1]; b2 = p[2]; b3 = p[3];
            }
            cbox[0] = b0; cbox[1] = b1; cbox[2] = b2; cbox[3] = b3;
            float* o = out + ((size_t)img * POSTN + k) * 5;
            o[0] = (idx >= 0) ? (float)img : 0.f;   // SCALE == 1.0
            o[1] = b0; o[2] = b1; o[3] = b2; o[4] = b3;
        }
        __syncthreads();
        const int idx = csel;
        if (idx >= 0) {
            const float ax1 = cbox[0], ay1 = cbox[1], ax2 = cbox[2], ay2 = cbox[3];
            const float area_a = (ax2 - ax1) * (ay2 - ay1);
            unsigned v = valid;
            while (v) {
                int j = __ffs(v) - 1;
                v &= v - 1;
                int i = j * 1024 + tid;
                const float* p = bx + (size_t)i * 4;
                float px1 = p[0], py1 = p[1], px2 = p[2], py2 = p[3];
                float ix1 = fmaxf(ax1, px1), iy1 = fmaxf(ay1, py1);
                float ix2 = fminf(ax2, px2), iy2 = fminf(ay2, py2);
                float inter = fmaxf(ix2 - ix1, 0.f) * fmaxf(iy2 - iy1, 0.f);
                float area_b = (px2 - px1) * (py2 - py1);
                float iou = inter / (area_a + area_b - inter);
                if (iou > 0.7f) valid &= ~(1u << j);
                if (i == idx)   valid &= ~(1u << j);
            }
        }
        __syncthreads();
    }
}

// ---------------------------------------------------------------------------
// Launcher
// ---------------------------------------------------------------------------
extern "C" void kernel_launch(void* const* d_in, const int* in_sizes, int n_in,
                              void* d_out, int out_size, void* d_ws, size_t ws_size,
                              hipStream_t stream) {
    const float* x   = (const float*)d_in[0];
    const float* cw  = (const float*)d_in[1];
    const float* cb  = (const float*)d_in[2];
    const float* sw  = (const float*)d_in[3];
    const float* sb  = (const float*)d_in[4];
    const float* bw  = (const float*)d_in[5];
    const float* bb  = (const float*)d_in[6];
    float* out = (float*)d_out;

    // workspace carve-out (256B aligned)
    char* ws = (char*)d_ws;
    size_t off = 0;
    auto carve = [&](size_t bytes) -> void* {
        void* p = ws + off;
        off = (off + bytes + 255) & ~(size_t)255;
        return p;
    };
    const size_t XPAD_E = (size_t)IMGS * HP * HP * CIN;   // bf16 elems
    const size_t WT_E   = (size_t)9 * COUT * CIN;         // bf16 elems
    unsigned short* xpad = (unsigned short*)carve(XPAD_E * 2);
    unsigned short* wt   = (unsigned short*)carve(WT_E * 2);
    float* hout   = (float*)carve((size_t)MTOT * COUT * 4);
    float* scores = (float*)carve((size_t)IMGS * NBOX * 4);
    float* boxes  = (float*)carve((size_t)IMGS * NBOX * 16);

    {
        int tot = (int)XPAD_E;
        pad_convert_x<<<(tot + 255) / 256, 256, 0, stream>>>(x, xpad);
    }
    {
        int tot = (int)WT_E;
        convert_w<<<(tot + 255) / 256, 256, 0, stream>>>(cw, wt);
    }
    {
        dim3 grid((MTOT + 127) / 128, COUT / 64);
        conv3x3_wmma<<<grid, 256, 0, stream>>>(xpad, wt, cb, hout);
    }
    heads_kernel<<<MTOT, 64, 0, stream>>>(hout, sw, sb, bw, bb, scores, boxes);
    nms_kernel<<<IMGS, 1024, 0, stream>>>(scores, boxes, out);
    (void)in_sizes; (void)n_in; (void)out_size; (void)ws_size;
}
